// MultiHeadCausalAttention_24111946400439
// MI455X (gfx1250) — matrix operand order repaired and checked by reading, not compiled, not run on hardware
//
#include <hip/hip_runtime.h>
#include <hip/hip_bf16.h>

typedef _Float16 f16;
typedef __attribute__((ext_vector_type(4)))  float    v4f;
typedef __attribute__((ext_vector_type(8)))  _Float16 v8h;
typedef __attribute__((ext_vector_type(16))) _Float16 v16h;
typedef __attribute__((ext_vector_type(8)))  float    v8f;
typedef int v4i_ __attribute__((vector_size(16)));   // matches builtin param pointee

#define DMODEL 1024
#define TSEQ   2048
#define BATCH  2
#define NHEAD  16
#define HDIM   64

static __device__ __forceinline__ v8f wmma_f16(v16h a, v16h b, v8f c) {
    // (neg_a, A, neg_b, B, c_mod, C, reuse_a, reuse_b)
    return __builtin_amdgcn_wmma_f32_16x16x32_f16(false, a, false, b, (short)0, c, false, false);
}

union V16U { v16h v; v8h h2[2]; f16 e[16]; };

// ---- async global->LDS copy of 16 bytes (gfx1250 GLOBAL_LOAD_ASYNC_TO_LDS_B128) ----
static __device__ __forceinline__ void async_cp16(f16* ldst, const f16* gsrc) {
#if __has_builtin(__builtin_amdgcn_global_load_async_to_lds_b128)
    __builtin_amdgcn_global_load_async_to_lds_b128(
        (__attribute__((address_space(1))) v4i_*)(size_t)gsrc,
        (__attribute__((address_space(3))) v4i_*)(unsigned)(size_t)ldst,
        0, 0);
#else
    asm volatile("global_load_async_to_lds_b128 %0, %1, off"
                 :: "v"((unsigned)(size_t)ldst),
                    "v"((unsigned long long)(size_t)gsrc)
                 : "memory");
#endif
}

static __device__ __forceinline__ void wait_async0() {
#if __has_builtin(__builtin_amdgcn_s_wait_asynccnt)
    __builtin_amdgcn_s_wait_asynccnt(0);
#else
    asm volatile("s_wait_asynccnt 0x0" ::: "memory");
#endif
}

// Build an f16 A-fragment from 16 fp32 values at xrow[kA..kA+7] and [kA+16..kA+23]
static __device__ __forceinline__ V16U load_a_frag_f32(const float* xrow, int kA) {
    v4f x0 = *(const v4f*)(xrow + kA);
    v4f x1 = *(const v4f*)(xrow + kA + 4);
    v4f x2 = *(const v4f*)(xrow + kA + 16);
    v4f x3 = *(const v4f*)(xrow + kA + 20);
    V16U a;
    #pragma unroll
    for (int e = 0; e < 4; ++e) {
        a.e[e]      = (f16)x0[e];
        a.e[4 + e]  = (f16)x1[e];
        a.e[8 + e]  = (f16)x2[e];
        a.e[12 + e] = (f16)x3[e];
    }
    return a;
}

// ---------------------------------------------------------------------------
// Kernel 1: fused QKV projection.  x(f32)[4096,1024] @ W(f32)[1024,1024] -> f16
// z=0 -> Q [bh][t][64], z=1 -> K [bh][t][64], z=2 -> V^T [bh][64][t]
// Block tile: 128 rows x 64 cols; each wave: two 16-row A tiles.
// ---------------------------------------------------------------------------
__global__ __launch_bounds__(128) void mha_qkv_proj(
    const float* __restrict__ x, const float* __restrict__ Wq,
    const float* __restrict__ Wk, const float* __restrict__ Wv,
    f16* __restrict__ ws)
{
    __shared__ __align__(16) f16 wt[64][72];   // W^T tile [n][k], padded stride

    const int n0  = blockIdx.x * 64;
    const int m0  = blockIdx.y * 128;
    const int z   = blockIdx.z;
    const float* W = (z == 0) ? Wq : (z == 1) ? Wk : Wv;
    f16* out = ws + (size_t)z * (size_t)BATCH * NHEAD * TSEQ * HDIM;

    const int tid  = threadIdx.x;
    const int wv   = tid >> 5;
    const int lane = tid & 31;
    const int lN   = lane & 15;
    const bool hi  = lane >= 16;

    v8f c[2][4] = {};
    const float* xrow0 = x + (size_t)(m0 + wv * 32 + lN) * DMODEL;
    const float* xrow1 = xrow0 + (size_t)16 * DMODEL;

    for (int kb = 0; kb < DMODEL; kb += 64) {
        __syncthreads();
        for (int i = tid; i < 64 * 64; i += 128) {
            int kk = i >> 6, nn = i & 63;
            wt[nn][kk] = (f16)W[(size_t)(kb + kk) * DMODEL + (n0 + nn)];
        }
        if (kb + 64 < DMODEL)
            __builtin_prefetch(&W[(size_t)(kb + 64 + (tid >> 1)) * DMODEL + n0 + (tid & 1) * 32], 0, 3);
        __syncthreads();

        #pragma unroll
        for (int ks = 0; ks < 2; ++ks) {
            const int kA = kb + ks * 32 + (hi ? 8 : 0);
            V16U a0 = load_a_frag_f32(xrow0, kA);
            V16U a1 = load_a_frag_f32(xrow1, kA);

            const int kB = ks * 32 + (hi ? 8 : 0);
            V16U bf[4];
            #pragma unroll
            for (int nt = 0; nt < 4; ++nt) {
                const f16* bp = &wt[nt * 16 + lN][kB];
                bf[nt].h2[0] = *(const v8h*)(bp);
                bf[nt].h2[1] = *(const v8h*)(bp + 16);
            }
            #pragma unroll
            for (int nt = 0; nt < 4; ++nt) {
                c[0][nt] = wmma_f16(a0.v, bf[nt].v, c[0][nt]);
                c[1][nt] = wmma_f16(a1.v, bf[nt].v, c[1][nt]);
            }
        }
    }

    #pragma unroll
    for (int mA = 0; mA < 2; ++mA) {
        #pragma unroll
        for (int nt = 0; nt < 4; ++nt) {
            const int n  = n0 + nt * 16 + lN;
            const int hh = n >> 6;
            const int d  = n & 63;
            #pragma unroll
            for (int r = 0; r < 8; ++r) {
                const int tok = m0 + wv * 32 + mA * 16 + r + (hi ? 8 : 0);
                const int bb = tok >> 11, t = tok & (TSEQ - 1);
                const f16 val = (f16)c[mA][nt][r];
                if (z < 2)
                    out[((size_t)(bb * NHEAD + hh) * TSEQ + t) * HDIM + d] = val;   // Q,K row-major
                else
                    out[((size_t)(bb * NHEAD + hh) * HDIM + d) * TSEQ + t] = val;   // V transposed
            }
        }
    }
}

// ---------------------------------------------------------------------------
// Kernel 2: causal flash attention.  One block = (b, h, 64-query tile).
// K/V^T tiles double-buffered in LDS via async global->LDS copies.
// ---------------------------------------------------------------------------
__global__ __launch_bounds__(128) void mha_flash_attn(
    const f16* __restrict__ Qw, const f16* __restrict__ Kw,
    const f16* __restrict__ Vt, f16* __restrict__ ctx)
{
    __shared__ __align__(16) f16 ktile[2][64][72];  // [buf][key][dim], padded
    __shared__ __align__(16) f16 vtile[2][64][72];  // [buf][dim][key], padded
    __shared__ __align__(16) f16 lp[4][16][72];     // per-wave P tile [16 q][64 k]

    const int qb  = blockIdx.x;
    const int h   = blockIdx.y;
    const int b   = blockIdx.z;
    const int bh  = b * NHEAD + h;
    const int tid = threadIdx.x;
    const int wv  = tid >> 5;
    const int lane = tid & 31;
    const int lN  = lane & 15;
    const bool hi = lane >= 16;
    const int q0  = qb * 64 + wv * 16;            // first query row of this wave

    const f16* krow_base = Kw + (size_t)bh * TSEQ * HDIM;
    const f16* vrow_base = Vt + (size_t)bh * HDIM * TSEQ;

    // cooperative async stage of one 64-key tile (K rows + V^T rows)
    auto stage_tile = [&](int kt_, int buf) {
        #pragma unroll
        for (int j = 0; j < 4; ++j) {
            const int i  = tid + j * 128;        // 0..511
            const int nn = i >> 3;
            const int cc = (i & 7) * 8;
            async_cp16(&ktile[buf][nn][cc], krow_base + (size_t)(kt_ * 64 + nn) * HDIM + cc);
            async_cp16(&vtile[buf][nn][cc], vrow_base + (size_t)nn * TSEQ + kt_ * 64 + cc);
        }
    };

    // Q A-fragments (2 k-steps over HDIM=64), kept in registers
    const f16* qrow = Qw + ((size_t)bh * TSEQ + (q0 + lN)) * HDIM;
    V16U qa[2];
    #pragma unroll
    for (int ks = 0; ks < 2; ++ks) {
        const int kA = ks * 32 + (hi ? 8 : 0);
        qa[ks].h2[0] = *(const v8h*)(qrow + kA);
        qa[ks].h2[1] = *(const v8h*)(qrow + kA + 16);
    }

    v8f o[4] = {};
    float mr[8], lr[8];
    #pragma unroll
    for (int r = 0; r < 8; ++r) { mr[r] = -__builtin_inff(); lr[r] = 0.f; }

    const float sscale = 0.125f * 1.4426950408889634f;   // 1/sqrt(64) * log2(e)

    stage_tile(0, 0);                              // prologue copy

    for (int kt = 0; kt <= qb; ++kt) {
        wait_async0();
        __syncthreads();                           // staged tile visible to all waves
        const int bufc = kt & 1;

        // ---- S = Q K^T for this 64-key tile (4 C-frags of 16x16) ----
        v8f s[4] = {};
        #pragma unroll
        for (int ks = 0; ks < 2; ++ks) {
            V16U bf[4];
            #pragma unroll
            for (int nt = 0; nt < 4; ++nt) {
                const f16* kp = &ktile[bufc][nt * 16 + lN][ks * 32 + (hi ? 8 : 0)];
                bf[nt].h2[0] = *(const v8h*)(kp);
                bf[nt].h2[1] = *(const v8h*)(kp + 16);
            }
            #pragma unroll
            for (int nt = 0; nt < 4; ++nt)
                s[nt] = wmma_f16(qa[ks].v, bf[nt].v, s[nt]);
        }

        // overlap next tile's copy with softmax + PV of this tile
        if (kt < qb) stage_tile(kt + 1, bufc ^ 1);

        // ---- causal mask (only on the diagonal tile) ----
        if (kt == qb) {
            #pragma unroll
            for (int nt = 0; nt < 4; ++nt) {
                const int key = kt * 64 + nt * 16 + lN;
                #pragma unroll
                for (int r = 0; r < 8; ++r) {
                    const int q = q0 + r + (hi ? 8 : 0);
                    if (key > q) s[nt][r] = -__builtin_inff();
                }
            }
        }

        // ---- online softmax (per query row M = r + 8*hi) ----
        #pragma unroll
        for (int r = 0; r < 8; ++r) {
            #pragma unroll
            for (int nt = 0; nt < 4; ++nt) s[nt][r] *= sscale;

            float rmax = fmaxf(fmaxf(s[0][r], s[1][r]), fmaxf(s[2][r], s[3][r]));
            rmax = fmaxf(rmax, __shfl_xor(rmax, 1, 32));
            rmax = fmaxf(rmax, __shfl_xor(rmax, 2, 32));
            rmax = fmaxf(rmax, __shfl_xor(rmax, 4, 32));
            rmax = fmaxf(rmax, __shfl_xor(rmax, 8, 32));

            const float mnew = fmaxf(mr[r], rmax);
            const float corr = exp2f(mr[r] - mnew);
            mr[r] = mnew;

            const int M = r + (hi ? 8 : 0);
            float rsum = 0.f;
            #pragma unroll
            for (int nt = 0; nt < 4; ++nt) {
                const float p = exp2f(s[nt][r] - mnew);
                rsum += p;
                lp[wv][M][nt * 16 + lN] = (f16)p;
            }
            rsum += __shfl_xor(rsum, 1, 32);
            rsum += __shfl_xor(rsum, 2, 32);
            rsum += __shfl_xor(rsum, 4, 32);
            rsum += __shfl_xor(rsum, 8, 32);

            lr[r] = lr[r] * corr + rsum;
            #pragma unroll
            for (int nt = 0; nt < 4; ++nt) o[nt][r] *= corr;
        }
        __syncthreads();

        // ---- O += P V  (A from LDS P tile, B from LDS V^T tile) ----
        #pragma unroll
        for (int ks = 0; ks < 2; ++ks) {
            const int kA = ks * 32 + (hi ? 8 : 0);
            V16U a;
            const f16* pr = &lp[wv][lN][kA];
            a.h2[0] = *(const v8h*)(pr);
            a.h2[1] = *(const v8h*)(pr + 16);

            V16U bf[4];
            #pragma unroll
            for (int nt = 0; nt < 4; ++nt) {
                const f16* vp = &vtile[bufc][nt * 16 + lN][ks * 32 + (hi ? 8 : 0)];
                bf[nt].h2[0] = *(const v8h*)(vp);
                bf[nt].h2[1] = *(const v8h*)(vp + 16);
            }
            #pragma unroll
            for (int nt = 0; nt < 4; ++nt)
                o[nt] = wmma_f16(a.v, bf[nt].v, o[nt]);
        }
        __syncthreads();                           // tile fully consumed before reuse
    }

    // ---- normalize and write ctx (f16, [tok][h*64+d]) ----
    #pragma unroll
    for (int r = 0; r < 8; ++r) {
        const float inv = 1.0f / lr[r];
        const size_t tok = (size_t)b * TSEQ + q0 + r + (hi ? 8 : 0);
        #pragma unroll
        for (int nt = 0; nt < 4; ++nt)
            ctx[tok * DMODEL + h * HDIM + nt * 16 + lN] = (f16)(o[nt][r] * inv);
    }
}

// ---------------------------------------------------------------------------
// Kernel 3: output projection  out = ctx @ Wo + bo   (f32 output)
// ---------------------------------------------------------------------------
__global__ __launch_bounds__(128) void mha_out_proj(
    const f16* __restrict__ ctx, const float* __restrict__ Wo,
    const float* __restrict__ bo, float* __restrict__ out)
{
    __shared__ __align__(16) f16 wt[64][72];

    const int n0  = blockIdx.x * 64;
    const int m0  = blockIdx.y * 128;
    const int tid = threadIdx.x;
    const int wv  = tid >> 5;
    const int lane = tid & 31;
    const int lN  = lane & 15;
    const bool hi = lane >= 16;

    v8f c[2][4] = {};
    const f16* arow0 = ctx + (size_t)(m0 + wv * 32 + lN) * DMODEL;
    const f16* arow1 = arow0 + (size_t)16 * DMODEL;

    for (int kb = 0; kb < DMODEL; kb += 64) {
        __syncthreads();
        for (int i = tid; i < 64 * 64; i += 128) {
            int kk = i >> 6, nn = i & 63;
            wt[nn][kk] = (f16)Wo[(size_t)(kb + kk) * DMODEL + (n0 + nn)];
        }
        if (kb + 64 < DMODEL)
            __builtin_prefetch(&Wo[(size_t)(kb + 64 + (tid >> 1)) * DMODEL + n0 + (tid & 1) * 32], 0, 3);
        __syncthreads();

        #pragma unroll
        for (int ks = 0; ks < 2; ++ks) {
            const int kA = kb + ks * 32 + (hi ? 8 : 0);
            V16U a0, a1;
            a0.h2[0] = *(const v8h*)(arow0 + kA);
            a0.h2[1] = *(const v8h*)(arow0 + kA + 16);
            a1.h2[0] = *(const v8h*)(arow1 + kA);
            a1.h2[1] = *(const v8h*)(arow1 + kA + 16);

            const int kB = ks * 32 + (hi ? 8 : 0);
            V16U bf[4];
            #pragma unroll
            for (int nt = 0; nt < 4; ++nt) {
                const f16* bp = &wt[nt * 16 + lN][kB];
                bf[nt].h2[0] = *(const v8h*)(bp);
                bf[nt].h2[1] = *(const v8h*)(bp + 16);
            }
            #pragma unroll
            for (int nt = 0; nt < 4; ++nt) {
                c[0][nt] = wmma_f16(a0.v, bf[nt].v, c[0][nt]);
                c[1][nt] = wmma_f16(a1.v, bf[nt].v, c[1][nt]);
            }
        }
    }

    #pragma unroll
    for (int mA = 0; mA < 2; ++mA) {
        #pragma unroll
        for (int nt = 0; nt < 4; ++nt) {
            const int n = n0 + nt * 16 + lN;
            const float bias = bo[n];
            #pragma unroll
            for (int r = 0; r < 8; ++r) {
                const size_t tok = (size_t)(m0 + wv * 32 + mA * 16 + r + (hi ? 8 : 0));
                out[tok * DMODEL + n] = c[mA][nt][r] + bias;
            }
        }
    }
}

// ---------------------------------------------------------------------------
extern "C" void kernel_launch(void* const* d_in, const int* in_sizes, int n_in,
                              void* d_out, int out_size, void* d_ws, size_t ws_size,
                              hipStream_t stream) {
    (void)in_sizes; (void)n_in; (void)out_size; (void)ws_size;
    const float* x  = (const float*)d_in[0];
    const float* Wq = (const float*)d_in[1];
    const float* Wk = (const float*)d_in[2];
    const float* Wv = (const float*)d_in[3];
    const float* Wo = (const float*)d_in[4];
    const float* bo = (const float*)d_in[5];
    float* out = (float*)d_out;

    const size_t per = (size_t)BATCH * NHEAD * TSEQ * HDIM;   // 4M halfs = 8 MB
    f16* ws  = (f16*)d_ws;
    f16* Qw  = ws;
    f16* Kw  = ws + per;
    f16* Vt  = ws + 2 * per;
    f16* ctx = ws + 3 * per;

    dim3 blk(128);
    mha_qkv_proj<<<dim3(DMODEL / 64, (BATCH * TSEQ) / 128, 3), blk, 0, stream>>>(x, Wq, Wk, Wv, ws);
    mha_flash_attn<<<dim3(TSEQ / 64, NHEAD, BATCH), blk, 0, stream>>>(Qw, Kw, Vt, ctx);
    mha_out_proj<<<dim3(DMODEL / 64, (BATCH * TSEQ) / 128), blk, 0, stream>>>(ctx, Wo, bo, out);
}
